// Semantic_Mapping_28501402976257
// MI455X (gfx1250) — compile-verified, matrix operand order checked by reading, not run on hardware
//
#include <hip/hip_runtime.h>
#include <math.h>

typedef __attribute__((ext_vector_type(16))) _Float16 v16h;
typedef __attribute__((ext_vector_type(8)))  float    v8f;

#define BSZ 4
#define CIN 20
#define HIN 480
#define WIN 640
#define VR 100
#define NZ 48
#define NF 17
#define MAPSZ 480
#define RADIUS 30
#define R2D 57.29577951308232f

// ---- workspace layout (floats) ----
#define VOX_OFF   ((size_t)0)             // 4*17*100*100*48 = 32,640,000
#define ALLP_OFF  ((size_t)32640000)      // 4*17*10000 = 680,000
#define AHP_OFF   ((size_t)33320000)
#define ASP_OFF   ((size_t)34000000)
#define ROT_OFF   ((size_t)34680000)      // 4*21*480*480 = 19,353,600
#define STAIR_OFF ((size_t)54033600)      // 4*480*480 = 921,600
#define PARAM_OFF ((size_t)54955200)      // 64 floats
#define ZERO_N    ((size_t)34680000)

// ---- output layout (floats) ----
#define O_TRANS   ((size_t)0)
#define O_MAPP    ((size_t)18432000)
#define O_MAPS    ((size_t)36864000)
#define O_POSE    ((size_t)55296000)

typedef __attribute__((address_space(3))) float* lds_fp;

__global__ void zero_kernel(float* p, size_t n) {
    size_t i = (size_t)blockIdx.x * blockDim.x + threadIdx.x;
    if (i < n) p[i] = 0.0f;
}

__global__ void pose_kernel(const float* __restrict__ pose_obs,
                            const float* __restrict__ poses_last,
                            float* __restrict__ params,
                            float* __restrict__ out_poses) {
    int b = threadIdx.x;
    if (b >= BSZ) return;
    float r = poses_last[b*3+2] / R2D;
    float po0 = pose_obs[b*3+0], po1 = pose_obs[b*3+1], po2 = pose_obs[b*3+2];
    float sr = sinf(r), cr = cosf(r);
    float ny = poses_last[b*3+1] + po0*sr + po1*cr;
    float nx = poses_last[b*3+0] + po0*cr - po1*sr;
    float nt = poses_last[b*3+2] + po2*R2D;
    nt = fmodf(nt - 180.0f, 360.0f) + 180.0f;
    nt = fmodf(nt + 180.0f, 360.0f) - 180.0f;
    out_poses[b*3+0] = nx; out_poses[b*3+1] = ny; out_poses[b*3+2] = nt;
    float sx = (240.0f - nx*20.0f) / 240.0f;   // 100/RES = 20
    float sy = (240.0f - ny*20.0f) / 240.0f;
    float tr = (90.0f - nt) * 0.017453292519943295f;
    params[b*8+0] = nx; params[b*8+1] = ny; params[b*8+2] = nt;
    params[b*8+3] = cosf(tr); params[b*8+4] = sinf(tr);
    params[b*8+5] = sx; params[b*8+6] = sy;
    if (b == 0) {
        int sxi = (int)(nx*20.0f); int syi = (int)(ny*20.0f);
        sxi = sxi < RADIUS ? RADIUS : (sxi > MAPSZ-RADIUS-1 ? MAPSZ-RADIUS-1 : sxi);
        syi = syi < RADIUS ? RADIUS : (syi > MAPSZ-RADIUS-1 ? MAPSZ-RADIUS-1 : syi);
        params[32] = (float)sxi; params[33] = (float)syi;
    }
}

// Point projection + trilinear scatter into voxel grid (atomic adds).
__global__ void splat_kernel(const float* __restrict__ obs,
                             const float* __restrict__ eve_angle,
                             float* __restrict__ vox) {
    int tid = blockIdx.x * blockDim.x + threadIdx.x;
    if (tid >= BSZ * HIN * WIN) return;
    int x = tid % WIN;
    int y = (tid / WIN) % HIN;
    int b = tid / (HIN * WIN);

    const float Ff = 320.0f / tanf(39.5f * 0.017453292519943295f);
    float depth = obs[(((size_t)b*CIN + 3)*HIN + y)*WIN + x];
    float gx = (float)x, gz = (float)(HIN - 1 - y);
    float X  = (gx - 319.5f) * depth / Ff;
    float Zc = (gz - 239.5f) * depth / Ff;
    float a  = eve_angle[b] * 0.017453292519943295f;
    float ca = cosf(a), sa = sinf(a);
    float Y  = ca*depth - sa*Zc;
    float Z  = sa*depth + ca*Zc + 88.0f;
    X += 250.0f;
    float xs = (X*0.2f - 50.0f) * 0.02f;
    float ys = (Y*0.2f - 50.0f) * 0.02f;
    float zs = (Z*0.2f - 16.0f) * (2.0f/48.0f);

    float p0 = xs*50.0f + 50.0f, p1 = ys*50.0f + 50.0f, p2 = zs*24.0f + 24.0f;
    float f0 = floorf(p0), f1 = floorf(p1), f2 = floorf(p2);
    float pd0[2], pd1[2], pd2[2], wd0[2], wd1[2], wd2[2];
    #pragma unroll
    for (int ix = 0; ix < 2; ++ix) {
        float q0 = f0 + ix; float s0 = (q0 > 0.0f && q0 < 100.0f) ? 1.0f : 0.0f;
        wd0[ix] = (1.0f - fabsf(p0 - q0)) * s0; pd0[ix] = q0 * s0;
        float q1 = f1 + ix; float s1 = (q1 > 0.0f && q1 < 100.0f) ? 1.0f : 0.0f;
        wd1[ix] = (1.0f - fabsf(p1 - q1)) * s1; pd1[ix] = q1 * s1;
        float q2 = f2 + ix; float s2 = (q2 > 0.0f && q2 < 48.0f) ? 1.0f : 0.0f;
        wd2[ix] = (1.0f - fabsf(p2 - q2)) * s2; pd2[ix] = q2 * s2;
    }
    float feat[NF];
    feat[0] = 1.0f;
    #pragma unroll
    for (int c = 1; c < NF; ++c)
        feat[c] = obs[(((size_t)b*CIN + 3 + c)*HIN + y)*WIN + x];

    #pragma unroll
    for (int i = 0; i < 2; ++i)
    #pragma unroll
    for (int j = 0; j < 2; ++j)
    #pragma unroll
    for (int k = 0; k < 2; ++k) {
        float w = wd0[i]*wd1[j]*wd2[k];
        if (w == 0.0f) continue;
        int idx = ((int)pd0[i]*VR + (int)pd1[j])*NZ + (int)pd2[k];
        #pragma unroll
        for (int c = 0; c < NF; ++c)
            atomicAdd(&vox[((size_t)b*NF + c)*480000 + idx], feat[c]*w);
    }
}

__device__ __forceinline__ float zmask(int n, int z) {
    if (z >= NZ) return 0.0f;
    if (n == 0) return 1.0f;                                   // allp
    if (n == 1) return (z >= 13 && z < 35) ? 1.0f : 0.0f;      // ahp  [MIN_Z,MAX_Z)
    if (n == 2) return (z >= 20 && z < 25) ? 1.0f : 0.0f;      // asp  [MID_Z-5,MID_Z)
    return 0.0f;
}

// Masked-matmul z-range reduction. Each wave:
//  1) stages its 16 voxel rows (16x48 f32 = 3KB) into LDS via coalesced
//     global_load_async_to_lds_b128 (6 issues x 32 lanes x 16B),
//  2) s_wait_asynccnt 0,
//  3) builds f16 A fragments from LDS (aligned float2 ds_loads) and 0/1
//     range-mask B fragments, runs two v_wmma_f32_16x16x32_f16
//     (K-chunks z[0:32), z[32:48) zero-padded) accumulating all three
//     range sums (columns 0..2) at once.
__global__ void zsum_wmma_kernel(const float* __restrict__ vox,
                                 float* __restrict__ allp,
                                 float* __restrict__ ahp,
                                 float* __restrict__ asp) {
    __shared__ float tile[8][16*NZ];     // 8 waves/block * 3KB = 24KB

    int tid  = blockIdx.x * blockDim.x + threadIdx.x;
    int wave = tid >> 5;
    int lane = threadIdx.x & 31;
    if (wave >= (BSZ*NF*VR*VR)/16) return;   // wave-uniform exit (EXEC all-1 otherwise)

    int basep = wave * 16;
    int wib   = threadIdx.x >> 5;            // wave-in-block
    float* twave = &tile[wib][0];

    // ---- stage 16 rows x 48 floats into LDS, coalesced 16B chunks ----
    #pragma unroll
    for (int i = 0; i < 6; ++i) {
        int c   = i*32 + lane;               // chunk id 0..191
        int r   = c / 12;                    // row 0..15
        int cik = c % 12;                    // 16B chunk within row
        int p   = basep + r;
        int xo  = p % VR;
        int yo  = (p / VR) % VR;
        int bc  = p / (VR*VR);
        const float* g = vox + (size_t)bc*480000 + (size_t)(xo*VR + yo)*NZ + cik*4;
        unsigned long long gaddr = (unsigned long long)g;
        unsigned ldsoff = (unsigned)(unsigned long long)(lds_fp)(twave + r*NZ + cik*4);
        asm volatile("global_load_async_to_lds_b128 %0, %1, off"
                     :: "v"(ldsoff), "v"(gaddr) : "memory");
    }
    asm volatile("s_wait_asynccnt 0x0" ::: "memory");

    int m    = lane & 15;
    int half = lane >> 4;
    const float* myrow = twave + m*NZ;

    v16h a0 = {}, a1 = {}, b0 = {}, b1 = {};
    #pragma unroll
    for (int v = 0; v < 8; ++v) {
        int k = (v < 4) ? ((half ? 8 : 0) + 2*v) : ((half ? 24 : 16) + 2*(v-4));
        float2 q0 = *(const float2*)(myrow + k);     // k even -> 8B aligned
        a0[2*v]   = (_Float16)rintf(q0.x);
        a0[2*v+1] = (_Float16)rintf(q0.y);
        int z1 = 32 + k;
        if (z1 < NZ) {
            float2 q1 = *(const float2*)(myrow + z1);
            a1[2*v]   = (_Float16)rintf(q1.x);
            a1[2*v+1] = (_Float16)rintf(q1.y);
        }
        b0[2*v]   = (_Float16)zmask(m, k);
        b0[2*v+1] = (_Float16)zmask(m, k+1);
        b1[2*v]   = (_Float16)zmask(m, z1);
        b1[2*v+1] = (_Float16)zmask(m, z1+1);
    }
    v8f acc = {};
    acc = __builtin_amdgcn_wmma_f32_16x16x32_f16(false, a0, false, b0,
                                                 (short)0, acc, false, false);
    acc = __builtin_amdgcn_wmma_f32_16x16x32_f16(false, a1, false, b1,
                                                 (short)0, acc, false, false);
    // D layout: lane column = lane%16; VGPR r -> row r + 8*(lane/16). Cols 0..2 used.
    if (m < 3) {
        float* outp = (m == 0) ? allp : (m == 1) ? ahp : asp;
        #pragma unroll
        for (int r = 0; r < 8; ++r)
            outp[basep + r + half*8] = acc[r];
    }
}

// agent_view computed on the fly: nonzero only in y[240,340) x[190,290)
__device__ __forceinline__ float agent_view_at(const float* __restrict__ allp,
                                               const float* __restrict__ ahp,
                                               const float* __restrict__ asp,
                                               int b, int ch, int iy, int ix) {
    if (iy < 240 || iy >= 340 || ix < 190 || ix >= 290) return 0.0f;
    int sp = ((b*NF)*VR + (iy-240))*VR + (ix-190);
    float v;
    if      (ch == 0)  v = ahp[sp];                    // fp_map   (MAP_T=1)
    else if (ch == 20) v = asp[sp];                    // fp_stair (MAP_T=1)
    else if (ch == 1)  v = allp[sp];                   // fp_exp   (EXP_T=1)
    else if (ch < 4)   return 0.0f;
    else               v = ahp[sp + (1 + ch - 4)*VR*VR] * 0.2f;   // /CAT_T
    return fminf(fmaxf(v, 0.0f), 1.0f);
}

__global__ void rotate_kernel(const float* __restrict__ allp,
                              const float* __restrict__ ahp,
                              const float* __restrict__ asp,
                              const float* __restrict__ params,
                              float* __restrict__ rotated) {
    size_t tid = (size_t)blockIdx.x * blockDim.x + threadIdx.x;
    if (tid >= (size_t)BSZ*21*MAPSZ*MAPSZ) return;
    int x  = tid % MAPSZ;
    int y  = (tid / MAPSZ) % MAPSZ;
    int ch = (tid / (MAPSZ*MAPSZ)) % 21;
    int b  = tid / ((size_t)21*MAPSZ*MAPSZ);
    float ct = params[b*8+3], st = params[b*8+4];
    float gx = (x + 0.5f)*(2.0f/MAPSZ) - 1.0f;
    float gy = (y + 0.5f)*(2.0f/MAPSZ) - 1.0f;
    float rx = ct*gx - st*gy, ry = st*gx + ct*gy;
    float px = (rx + 1.0f)*0.5f*(MAPSZ-1), py = (ry + 1.0f)*0.5f*(MAPSZ-1);
    float fx = floorf(px), fy = floorf(py);
    int x0 = (int)fx, y0 = (int)fy;
    float wx = px - fx, wy = py - fy;
    float v00 = agent_view_at(allp, ahp, asp, b, ch, y0,   x0);
    float v01 = agent_view_at(allp, ahp, asp, b, ch, y0,   x0+1);
    float v10 = agent_view_at(allp, ahp, asp, b, ch, y0+1, x0);
    float v11 = agent_view_at(allp, ahp, asp, b, ch, y0+1, x0+1);
    rotated[tid] = v00*(1.0f-wx)*(1.0f-wy) + v01*wx*(1.0f-wy)
                 + v10*(1.0f-wx)*wy       + v11*wx*wy;
}

__device__ __forceinline__ float samp480(const float* __restrict__ pl, int iy, int ix) {
    if (iy < 0 || iy >= MAPSZ || ix < 0 || ix >= MAPSZ) return 0.0f;
    return pl[iy*MAPSZ + ix];
}

__global__ void translate_kernel(const float* __restrict__ rotated,
                                 const float* __restrict__ params,
                                 float* __restrict__ out_trans,
                                 float* __restrict__ stair_buf) {
    size_t tid = (size_t)blockIdx.x * blockDim.x + threadIdx.x;
    if (tid >= (size_t)BSZ*21*MAPSZ*MAPSZ) return;
    int x  = tid % MAPSZ;
    int y  = (tid / MAPSZ) % MAPSZ;
    int ch = (tid / (MAPSZ*MAPSZ)) % 21;
    int b  = tid / ((size_t)21*MAPSZ*MAPSZ);
    float sx = params[b*8+5], sy = params[b*8+6];
    float gx = (x + 0.5f)*(2.0f/MAPSZ) - 1.0f + sx;
    float gy = (y + 0.5f)*(2.0f/MAPSZ) - 1.0f + sy;
    float px = (gx + 1.0f)*0.5f*(MAPSZ-1), py = (gy + 1.0f)*0.5f*(MAPSZ-1);
    float fx = floorf(px), fy = floorf(py);
    int x0 = (int)fx, y0 = (int)fy;
    float wx = px - fx, wy = py - fy;
    const float* pl = rotated + ((size_t)(b*21 + ch))*MAPSZ*MAPSZ;
    int ypf = y0 < 0 ? 0 : (y0 >= MAPSZ ? MAPSZ-1 : y0);
    __builtin_prefetch(pl + ypf*MAPSZ, 0, 1);     // gfx1250 global_prefetch_b8
    float v00 = samp480(pl, y0,   x0);
    float v01 = samp480(pl, y0,   x0+1);
    float v10 = samp480(pl, y0+1, x0);
    float v11 = samp480(pl, y0+1, x0+1);
    float out = v00*(1.0f-wx)*(1.0f-wy) + v01*wx*(1.0f-wy)
              + v10*(1.0f-wx)*wy       + v11*wx*wy;
    if (ch < CIN)
        out_trans[(((size_t)b*CIN + ch)*MAPSZ + y)*MAPSZ + x] = out;
    else
        stair_buf[((size_t)b*MAPSZ + y)*MAPSZ + x] = out;
}

__device__ __forceinline__ float stair_grid_at(int y, int x, int syi, int sxi) {
    int i = y - (syi - RADIUS), j = x - (sxi - RADIUS);
    if (i < 0 || i >= 2*RADIUS || j < 0 || j >= 2*RADIUS) return 0.0f;
    float di = i + 0.5f - RADIUS, dj = j + 0.5f - RADIUS;
    return (di*di + dj*dj <= (float)(RADIUS*RADIUS)) ? 1.0f : 0.0f;
}

__global__ void finalize_kernel(const float* __restrict__ trans,
                                const float* __restrict__ stair_buf,
                                const float* __restrict__ maps_last,
                                const float* __restrict__ eve_angle,
                                const float* __restrict__ params,
                                float* __restrict__ map_pred,
                                float* __restrict__ map_pred_stair) {
    size_t tid = (size_t)blockIdx.x * blockDim.x + threadIdx.x;
    if (tid >= (size_t)BSZ*CIN*MAPSZ*MAPSZ) return;
    int x  = tid % MAPSZ;
    int y  = (tid / MAPSZ) % MAPSZ;
    int ch = (tid / (MAPSZ*MAPSZ)) % CIN;
    int b  = tid / ((size_t)CIN*MAPSZ*MAPSZ);

    float t   = trans[tid];
    float ml  = maps_last[tid];
    float eve = eve_angle[b];
    int sxi = (int)params[32], syi = (int)params[33];
    float sgv = stair_grid_at(y, x, syi, sxi);

    // map_pred
    float mp = fmaxf(ml, t);
    if (ch == 0) {
        const float* c0 = trans + ((size_t)b*CIN)*MAPSZ*MAPSZ;
        float mp3 = -INFINITY;
        #pragma unroll
        for (int dy = -1; dy <= 1; ++dy)
        #pragma unroll
        for (int dx = -1; dx <= 1; ++dx) {
            int yy = y + dy, xx = x + dx;
            if (yy >= 0 && yy < MAPSZ && xx >= 0 && xx < MAPSZ)
                mp3 = fmaxf(mp3, c0[yy*MAPSZ + xx]);
        }
        float d = trans[(((size_t)b*CIN + 1)*MAPSZ + y)*MAPSZ + x] - mp3;
        if (d > 0.8f && eve == 0.0f) mp = 0.0f;
    }
    map_pred[tid] = mp;

    // map_pred_stair
    float ts = (ch == 0) ? stair_buf[((size_t)b*MAPSZ + y)*MAPSZ + x] : t;
    if (b == 0 && ch < 2) ts *= sgv;
    float mps = fmaxf(ml, ts);
    if (ch == 0) {
        float ts1 = trans[(((size_t)b*CIN + 1)*MAPSZ + y)*MAPSZ + x];
        if (b == 0) ts1 *= sgv;
        float ds = ts1 - ts;
        if (ds > 0.8f && eve == 0.0f) mps = 0.0f;
    }
    map_pred_stair[tid] = mps;
}

extern "C" void kernel_launch(void* const* d_in, const int* in_sizes, int n_in,
                              void* d_out, int out_size, void* d_ws, size_t ws_size,
                              hipStream_t stream) {
    const float* obs        = (const float*)d_in[0];
    const float* pose_obs   = (const float*)d_in[1];
    const float* maps_last  = (const float*)d_in[2];
    const float* poses_last = (const float*)d_in[3];
    const float* eve_angle  = (const float*)d_in[4];
    float* out = (float*)d_out;
    float* ws  = (float*)d_ws;

    float* vox       = ws + VOX_OFF;
    float* allp      = ws + ALLP_OFF;
    float* ahp       = ws + AHP_OFF;
    float* asp       = ws + ASP_OFF;
    float* rotated   = ws + ROT_OFF;
    float* stair_buf = ws + STAIR_OFF;
    float* params    = ws + PARAM_OFF;

    const int TB = 256;

    zero_kernel<<<(int)((ZERO_N + TB - 1)/TB), TB, 0, stream>>>(ws, ZERO_N);
    pose_kernel<<<1, 32, 0, stream>>>(pose_obs, poses_last, params, out + O_POSE);

    int npix = BSZ * HIN * WIN;                       // 1,228,800
    splat_kernel<<<(npix + TB - 1)/TB, TB, 0, stream>>>(obs, eve_angle, vox);

    int nzsum_threads = (BSZ*NF*VR*VR/16) * 32;       // 42,500 waves * 32
    zsum_wmma_kernel<<<(nzsum_threads + TB - 1)/TB, TB, 0, stream>>>(vox, allp, ahp, asp);

    size_t nwarp = (size_t)BSZ*21*MAPSZ*MAPSZ;        // 19,353,600
    rotate_kernel<<<(int)((nwarp + TB - 1)/TB), TB, 0, stream>>>(allp, ahp, asp, params, rotated);
    translate_kernel<<<(int)((nwarp + TB - 1)/TB), TB, 0, stream>>>(rotated, params,
                                                                    out + O_TRANS, stair_buf);

    size_t nmap = (size_t)BSZ*CIN*MAPSZ*MAPSZ;        // 18,432,000
    finalize_kernel<<<(int)((nmap + TB - 1)/TB), TB, 0, stream>>>(out + O_TRANS, stair_buf,
                                                                  maps_last, eve_angle, params,
                                                                  out + O_MAPP, out + O_MAPS);
}
